// C_Cross_Attention3D_64458869178985
// MI455X (gfx1250) — matrix operand order repaired and checked by reading, not compiled, not run on hardware
//
#include <hip/hip_runtime.h>
#include <hip/hip_bf16.h>

#define HEADS 12
#define Bn    16
#define Cn    768
#define Nn    4096
#define HD    64
#define TT    128                 // token tile per block
#define CHUNKS (Nn / TT)          // 32
#define KC    128                 // K (channel) chunk staged per head
#define NKC   (Cn / KC)           // 6
#define PA    776                 // Abuf pitch (bf16 elems), multiple of 8 -> 16B rows
#define PW    144                 // Wbuf pitch (bf16 elems), multiple of 8

typedef __bf16 v16bf __attribute__((ext_vector_type(16)));
typedef float  v8f   __attribute__((ext_vector_type(8)));
typedef int    i4    __attribute__((ext_vector_type(4)));   // trivially-copyable 16B
typedef float  f4    __attribute__((ext_vector_type(4)));
typedef __attribute__((address_space(1))) i4 as1_i4;        // global-AS v4i32
typedef __attribute__((address_space(3))) i4 as3_i4;        // LDS-AS v4i32

// CDNA5 async global->LDS staging (guarded: falls back to sync copy if the
// toolchain doesn't expose the builtins; histogram will show which path won)
#if defined(__has_builtin)
#if __has_builtin(__builtin_amdgcn_global_load_async_to_lds_b128) && \
    __has_builtin(__builtin_amdgcn_s_wait_asynccnt)
#define CA3D_ASYNC 1
#endif
#endif
#ifndef CA3D_ASYNC
#define CA3D_ASYNC 0
#endif

#if CA3D_ASYNC
#define CA3D_ASYNC_WAIT() __builtin_amdgcn_s_wait_asynccnt(0)
#else
#define CA3D_ASYNC_WAIT()
#endif

// ---- workspace layout (bytes); total ~4.1 MB ----
#define WS_Q    0                         // q_norm: 16*12*64 f32          = 49152 B
#define WS_WHV  (49152)                   // Whv bf16: 12*128*768          = 2359296 B
#define WS_BKV  (WS_WHV + 2359296)        // permuted bkv: 12*128 f32      = 6144 B
#define WS_PART (WS_BKV + 6144)           // partials: 16*12*32*66 f32     = 1622016 B
#define WS_ATTN (WS_PART + 1622016)       // attn out: 16*768 f32          = 49152 B

// ---------------- P0: permute Wkv -> [head][col(k|v)][C] bf16, permute bias ----------------
__global__ void ca3d_prep_w(const float* __restrict__ Wkv, const float* __restrict__ bkv,
                            unsigned int* __restrict__ Whv32, float* __restrict__ bkvp) {
  const int PER = 128 * (Cn / 2);                       // bf16-pairs per head
  int e = blockIdx.x * blockDim.x + threadIdx.x;
  if (e < HEADS * PER) {
    int h = e / PER;
    int rem = e - h * PER;
    int j  = rem / (Cn / 2);
    int c2 = rem - j * (Cn / 2);
    int srow = (j < HD) ? (h * HD + j) : (Cn + h * HD + (j - HD));
    float f0 = Wkv[(size_t)srow * Cn + c2 * 2 + 0];
    float f1 = Wkv[(size_t)srow * Cn + c2 * 2 + 1];
    union { __bf16 b[2]; unsigned int u; } pk;
    pk.b[0] = (__bf16)f0;
    pk.b[1] = (__bf16)f1;
    Whv32[(size_t)(h * 128 + j) * (Cn / 2) + c2] = pk.u;
  }
  if (e < HEADS * 128) {
    int h = e >> 7, j = e & 127;
    int srow = (j < HD) ? (h * HD + j) : (Cn + h * HD + (j - HD));
    bkvp[e] = bkv[srow];
  }
}

// ---------------- P1: q = y Wq^T + bq, per-head L2 normalize ----------------
__global__ void ca3d_qproj(const float* __restrict__ y, const float* __restrict__ Wq,
                           const float* __restrict__ bq, float* __restrict__ qn) {
  int bh = blockIdx.x;                    // b*HEADS + h
  int b = bh / HEADS, h = bh - b * HEADS;
  int d = threadIdx.x;                    // 0..63
  int o = h * HD + d;
  const float* yb = y + (size_t)b * Cn;
  const float* wr = Wq + (size_t)o * Cn;
  float acc = bq[o];
  for (int c = 0; c < Cn; ++c) acc += yb[c] * wr[c];
  __shared__ float sh[HD];
  sh[d] = acc * acc;
  __syncthreads();
  float ssq = 0.f;
  for (int i = 0; i < HD; ++i) ssq += sh[i];
  qn[(size_t)bh * HD + d] = acc / fmaxf(sqrtf(ssq), 1e-12f);
}

// stage one 128-col x KC weight chunk into LDS (async if available)
__device__ __forceinline__ void ca3d_stage_w(__bf16* dst, const unsigned int* Whv32,
                                             int h, int kc, int tid) {
  const char* gbase = (const char*)Whv32;
  for (int idx = tid; idx < (128 * KC) / 8; idx += 256) {   // 16B units: 2048
    int j   = idx >> 4;                                     // row (col of B)
    int seg = idx & 15;                                     // 16B segment in row
    const char* g = gbase + ((size_t)(h * 128 + j) * Cn + kc * KC) * 2 + seg * 16;
#if CA3D_ASYNC
    __builtin_amdgcn_global_load_async_to_lds_b128(
        (as1_i4*)(uintptr_t)g,
        (as3_i4*)(unsigned int)(uintptr_t)&dst[j * PW + seg * 8],
        0, 0);
#else
    *(i4*)&dst[j * PW + seg * 8] = *(const i4*)g;
#endif
  }
}

// ---------------- Main: fused kv-GEMM (bf16 WMMA) + cosine attention partials ----------------
__global__ __launch_bounds__(256, 1)
void ca3d_attn_main(const float* __restrict__ x, const unsigned int* __restrict__ Whv32,
                    const float* __restrict__ bkvp, const float* __restrict__ qn,
                    float* __restrict__ part) {
  __shared__ __bf16 Abuf[TT * PA];        // 198656 B  x-tile, token-major, K contiguous
  __shared__ __bf16 Wbuf[2][128 * PW];    // 2x36864 B double-buffered weight chunk
  __shared__ float  qbuf[Cn];
  __shared__ float  Obuf[HD];
  __shared__ float  redMax[8];
  __shared__ float  Psh;

  const int tid   = threadIdx.x;
  const int blk   = blockIdx.x;
  const int b     = blk / CHUNKS;
  const int chunk = blk - b * CHUNKS;
  const int n0    = chunk * TT;

  for (int i = tid; i < Cn; i += 256) qbuf[i] = qn[(size_t)b * Cn + i];

  // stage x tile: x[b, c, n0..n0+127] -> Abuf[t][c] (f32 -> bf16, transposed)
  const float* xb = x + (size_t)b * Cn * Nn + n0;
  for (int idx = tid; idx < Cn * (TT / 4); idx += 256) {
    int c  = idx >> 5;                    // TT/4 == 32
    int t4 = idx & 31;
    f4 xv = *(const f4*)(xb + (size_t)c * Nn + t4 * 4);
    int t = t4 * 4;
    Abuf[(t + 0) * PA + c] = (__bf16)xv.x;
    Abuf[(t + 1) * PA + c] = (__bf16)xv.y;
    Abuf[(t + 2) * PA + c] = (__bf16)xv.z;
    Abuf[(t + 3) * PA + c] = (__bf16)xv.w;
  }

  const int lane = tid & 31;
  const int w    = tid >> 5;              // wave 0..7 -> M tile
  const int half = lane >> 4;
  const int ln   = lane & 15;
  const int m0   = w * 16;

  // kick off first weight chunk while Abuf staging is still in flight
  ca3d_stage_w(Wbuf[0], Whv32, 0, 0, tid);

  for (int h = 0; h < HEADS; ++h) {
    v8f acc[8];
    #pragma unroll
    for (int j = 0; j < 8; ++j) acc[j] = (v8f){0.f,0.f,0.f,0.f,0.f,0.f,0.f,0.f};

    for (int kc = 0; kc < NKC; ++kc) {
      CA3D_ASYNC_WAIT();                  // own async stage(kc) complete
      __syncthreads();                    // block: stage(kc) visible, compute(kc-1) done
      if (kc + 1 < NKC)        ca3d_stage_w(Wbuf[(kc + 1) & 1], Whv32, h, kc + 1, tid);
      else if (h + 1 < HEADS)  ca3d_stage_w(Wbuf[0],            Whv32, h + 1, 0,  tid);

      const __bf16* Wb = Wbuf[kc & 1];
      #pragma unroll
      for (int ks = 0; ks < KC / 32; ++ks) {
        union F { i4 q[2]; v16bf v; } ua, bc, bn;
        // A fragment (16x32 bf16): lanes 0-15 K {kb..kb+15}, lanes 16-31 K {kb+16..kb+31}
        const i4* pa = (const i4*)&Abuf[(m0 + ln) * PA + kc * KC + ks * 32 + half * 16];
        ua.q[0] = pa[0];
        ua.q[1] = pa[1];                  // +8 elems = +16B
        {
          const i4* pb = (const i4*)&Wb[ln * PW + ks * 32 + half * 16];
          bc.q[0] = pb[0]; bc.q[1] = pb[1];
        }
        #pragma unroll
        for (int j = 0; j < 8; ++j) {
          if (j < 7) {                    // prefetch next column tile's B fragment
            const i4* pb = (const i4*)&Wb[((j + 1) * 16 + ln) * PW + ks * 32 + half * 16];
            bn.q[0] = pb[0]; bn.q[1] = pb[1];
          }
          acc[j] = __builtin_amdgcn_wmma_f32_16x16x32_bf16(
              false, ua.v, false, bc.v, (short)0, acc[j], false, false);
          if (j < 7) bc = bn;
        }
      }
    }

    // epilogue: bias, k-norm, score, block-local stable softmax, weighted V
    float bias[8];
    #pragma unroll
    for (int j = 0; j < 8; ++j) bias[j] = bkvp[h * 128 + j * 16 + ln];
    #pragma unroll
    for (int j = 0; j < 8; ++j)
      #pragma unroll
      for (int i = 0; i < 8; ++i) acc[j][i] += bias[j];

    float qv[4];
    #pragma unroll
    for (int j = 0; j < 4; ++j) qv[j] = qbuf[h * HD + j * 16 + ln];

    float sv[8];
    float smax = -3.0e38f;
    #pragma unroll
    for (int i = 0; i < 8; ++i) {        // token t = m0 + half*8 + i
      float ssq = 0.f, dot = 0.f;
      #pragma unroll
      for (int j = 0; j < 4; ++j) { float k = acc[j][i]; ssq += k * k; dot += k * qv[j]; }
      #pragma unroll
      for (int m = 1; m <= 8; m <<= 1) { // reduce 64 cols across 16 lanes
        ssq += __shfl_xor(ssq, m, 32);
        dot += __shfl_xor(dot, m, 32);
      }
      float s = dot / fmaxf(sqrtf(ssq), 1e-12f);
      sv[i] = s;
      smax = fmaxf(smax, s);
    }
    smax = fmaxf(smax, __shfl_xor(smax, 16, 32));

    if (lane == 0) redMax[w] = smax;
    if (tid < HD)  Obuf[tid] = 0.f;
    if (tid == 64) Psh = 0.f;
    __syncthreads();
    float Mloc = redMax[0];
    #pragma unroll
    for (int i = 1; i < 8; ++i) Mloc = fmaxf(Mloc, redMax[i]);

    float ev[8], pl = 0.f;
    #pragma unroll
    for (int i = 0; i < 8; ++i) { ev[i] = __expf(sv[i] - Mloc); pl += ev[i]; }

    #pragma unroll
    for (int j = 4; j < 8; ++j) {        // v columns
      float val = 0.f;
      #pragma unroll
      for (int i = 0; i < 8; ++i) val += ev[i] * acc[j][i];
      val += __shfl_xor(val, 16, 32);    // combine both halves' tokens
      if (half == 0) atomicAdd(&Obuf[(j - 4) * 16 + ln], val);   // ds_add_f32
    }
    pl += __shfl_xor(pl, 16, 32);
    if (lane == 0) atomicAdd(&Psh, pl);
    __syncthreads();

    size_t base = (size_t)((b * HEADS + h) * CHUNKS + chunk) * 66;
    if (tid < HD)  part[base + 2 + tid] = Obuf[tid];
    if (tid == 64) part[base + 0] = Mloc;
    if (tid == 65) part[base + 1] = Psh;
    __syncthreads();
  }
}

// ---------------- Reduce chunk partials -> attn output per (b,h) ----------------
__global__ void ca3d_reduce(const float* __restrict__ part, float* __restrict__ attn) {
  int bh = blockIdx.x;                    // 0..191
  int n  = threadIdx.x;                   // 0..63
  size_t base = (size_t)bh * CHUNKS * 66;
  float M = -3.0e38f;
  for (int c = 0; c < CHUNKS; ++c) M = fmaxf(M, part[base + (size_t)c * 66]);
  float P = 0.f, O = 0.f;
  for (int c = 0; c < CHUNKS; ++c) {
    float sc = __expf(part[base + (size_t)c * 66] - M);
    P += part[base + (size_t)c * 66 + 1] * sc;
    O += part[base + (size_t)c * 66 + 2 + n] * sc;
  }
  int b = bh / HEADS, h = bh - b * HEADS;
  attn[(size_t)b * Cn + h * HD + n] = O / P;
}

// ---------------- Final projection: out = attn Wp^T + bp ----------------
__global__ void ca3d_proj(const float* __restrict__ attn, const float* __restrict__ Wp,
                          const float* __restrict__ bp, float* __restrict__ out) {
  __shared__ float arow[Cn];
  int b  = blockIdx.x / 3;
  int og = (blockIdx.x - b * 3) * 256 + threadIdx.x;
  for (int i = threadIdx.x; i < Cn; i += 256) arow[i] = attn[(size_t)b * Cn + i];
  __syncthreads();
  const float* wr = Wp + (size_t)og * Cn;
  float acc = bp[og];
  for (int c = 0; c < Cn; ++c) acc += arow[c] * wr[c];
  out[(size_t)b * Cn + og] = acc;
}

extern "C" void kernel_launch(void* const* d_in, const int* in_sizes, int n_in,
                              void* d_out, int out_size, void* d_ws, size_t ws_size,
                              hipStream_t stream) {
  const float* x   = (const float*)d_in[0];
  const float* y   = (const float*)d_in[1];
  const float* Wq  = (const float*)d_in[2];
  const float* bq  = (const float*)d_in[3];
  const float* Wkv = (const float*)d_in[4];
  const float* bkv = (const float*)d_in[5];
  const float* Wp  = (const float*)d_in[6];
  const float* bp  = (const float*)d_in[7];
  float* out = (float*)d_out;

  char* ws = (char*)d_ws;
  float*        qnrm  = (float*)(ws + WS_Q);
  unsigned int* Whv32 = (unsigned int*)(ws + WS_WHV);
  float*        bkvp  = (float*)(ws + WS_BKV);
  float*        partb = (float*)(ws + WS_PART);
  float*        attnb = (float*)(ws + WS_ATTN);

  ca3d_prep_w<<<(HEADS * 128 * (Cn / 2) + 255) / 256, 256, 0, stream>>>(Wkv, bkv, Whv32, bkvp);
  ca3d_qproj<<<Bn * HEADS, HD, 0, stream>>>(y, Wq, bq, qnrm);
  ca3d_attn_main<<<Bn * CHUNKS, 256, 0, stream>>>(x, Whv32, bkvp, qnrm, partb);
  ca3d_reduce<<<Bn * HEADS, HD, 0, stream>>>(partb, attnb);
  ca3d_proj<<<Bn * 3, 256, 0, stream>>>(attnb, Wp, bp, out);
}
